// CausalSelfAttention_10505490006057
// MI455X (gfx1250) — compile-verified, matrix operand order checked by reading, not run on hardware
//
#include <hip/hip_runtime.h>

// ---------------------------------------------------------------------------
// Causal self-attention, MI455X (gfx1250), bf16 WMMA with f32 accumulation.
// B=4, T=2048, C=1024, H=16, D=64.
// ---------------------------------------------------------------------------

#define BATCH 4
#define TLEN  2048
#define CDIM  1024
#define HEADS 16
#define DHEAD 64

typedef __attribute__((ext_vector_type(16))) __bf16 bf16x16;
typedef __attribute__((ext_vector_type(8)))  float  f32x8;

union Frag16 {
    bf16x16 v;
    uint4   q[2];
    unsigned short s[16];
};

// float -> bf16, round-to-nearest-even
__device__ __forceinline__ unsigned short f2bf(float x) {
    unsigned u = __float_as_uint(x);
    unsigned r = u + 0x7FFFu + ((u >> 16) & 1u);
    return (unsigned short)(r >> 16);
}

// Pointer to this lane's first 8-element chunk of a 16x32 bf16 fragment.
// CDNA5 16-bit A-matrix layout: lanes 0-15 hold row (r0+lane), K {0..7};
// lanes 16-31 hold row (r0+lane-16), K {8..15}; second 16B load (p+16)
// covers K {16..23} / {24..31}.
__device__ __forceinline__ const unsigned short*
frag_ptr(const unsigned short* base, int ld, int r0, int k0, int lane) {
    return base + (size_t)(r0 + (lane & 15)) * ld + k0 + (lane >> 4) * 8;
}

__device__ __forceinline__ bf16x16 load_frag(const unsigned short* p) {
    Frag16 f;
    f.q[0] = *(const uint4*)(p);
    f.q[1] = *(const uint4*)(p + 16);
    return f.v;
}

__device__ __forceinline__ f32x8 wmma_bf16(bf16x16 a, bf16x16 b, f32x8 c) {
    return __builtin_amdgcn_wmma_f32_16x16x32_bf16(
        /*neg_a=*/false, a, /*neg_b=*/false, b,
        /*c_mod=*/(short)0, c, /*reuse_a=*/false, /*reuse_b=*/false);
}

// ---------------------------------------------------------------------------
// Conversion kernels
// ---------------------------------------------------------------------------

__global__ void convert_f32_to_bf16(const float* __restrict__ in,
                                    unsigned short* __restrict__ out, int n) {
    int gid = blockIdx.x * blockDim.x + threadIdx.x;
    if (gid < n) out[gid] = f2bf(in[gid]);
}

// W [K=1024][N=1024] f32 row-major -> WT [N][K] bf16 row-major
__global__ void convert_transpose_w(const float* __restrict__ W,
                                    unsigned short* __restrict__ WT) {
    int gid = blockIdx.x * blockDim.x + threadIdx.x;   // 1024*1024 threads
    int k = gid >> 10;
    int n = gid & 1023;
    WT[n * CDIM + k] = f2bf(W[gid]);
}

// ---------------------------------------------------------------------------
// WMMA GEMM: C[M,N] = A[M,K](bf16) * BT[N,K]^T(bf16) + bias
// Register-blocked: one wave computes a 32x64 output tile (2 M-subtiles x
// 4 N-subtiles = 8 f32x8 accumulators). Per K=32 step: 2 A-frags are reused
// across 4 B-frags -> 8 WMMAs per 12 global_load_b128.
// mode 0: write f32 row-major C (final projection -> d_out)
// mode 1: write bf16 [B,H,T,D]   (q, k)
// mode 2: write bf16 [B,H,D,T]   (v transposed)
// grid = (N/64, M/32), block = 32 (one wave).
// ---------------------------------------------------------------------------

__global__ void gemm_wmma(const unsigned short* __restrict__ A,
                          const unsigned short* __restrict__ BT,
                          const float* __restrict__ bias,
                          void* __restrict__ out,
                          int M, int N, int K, int mode) {
    const int lane = threadIdx.x;
    const int n0 = blockIdx.x * 64;
    const int m0 = blockIdx.y * 32;

    f32x8 acc[2][4];
#pragma unroll
    for (int mi = 0; mi < 2; ++mi)
#pragma unroll
        for (int c = 0; c < 4; ++c) acc[mi][c] = (f32x8){};

#pragma unroll 2
    for (int k0 = 0; k0 < K; k0 += 32) {
        // prefetch fragment lines 4 K-blocks ahead (k0+128) into L2/L0
        __builtin_prefetch(frag_ptr(A,  K, m0,      k0 + 128, lane), 0, 1);
        __builtin_prefetch(frag_ptr(BT, K, n0,      k0 + 128, lane), 0, 1);

        bf16x16 a0 = load_frag(frag_ptr(A, K, m0,      k0, lane));
        bf16x16 a1 = load_frag(frag_ptr(A, K, m0 + 16, k0, lane));
#pragma unroll
        for (int c = 0; c < 4; ++c) {
            bf16x16 b = load_frag(frag_ptr(BT, K, n0 + c * 16, k0, lane));
            acc[0][c] = wmma_bf16(a0, b, acc[0][c]);
            acc[1][c] = wmma_bf16(a1, b, acc[1][c]);
        }
    }

    // C layout: VGPR i -> row (subtile_m0 + i + 8*(lane>=16)), col (lane&15)
    const int lcol = lane & 15;
    const int hi8  = 8 * (lane >> 4);

#pragma unroll
    for (int mi = 0; mi < 2; ++mi) {
#pragma unroll
        for (int c = 0; c < 4; ++c) {
            const int col = n0 + c * 16 + lcol;
            const int rbase = m0 + mi * 16 + hi8;
            const float bv = bias[col];
#pragma unroll
            for (int i = 0; i < 8; ++i) {
                float val = acc[mi][c][i] + bv;
                int row = rbase + i;
                if (mode == 0) {
                    ((float*)out)[(size_t)row * N + col] = val;
                } else {
                    int b_ = row >> 11;          // T = 2048
                    int t  = row & (TLEN - 1);
                    int h  = col >> 6;           // D = 64
                    int d  = col & (DHEAD - 1);
                    unsigned short bf = f2bf(val);
                    if (mode == 1) {
                        ((unsigned short*)out)[(((size_t)(b_ * HEADS + h) * TLEN + t) * DHEAD) + d] = bf;
                    } else {
                        ((unsigned short*)out)[(((size_t)(b_ * HEADS + h) * DHEAD + d) * TLEN) + t] = bf;
                    }
                }
            }
        }
    }
}

// ---------------------------------------------------------------------------
// Flash attention core. One wave per (b*h, 16-query-row tile).
// q,k: bf16 [B,H,T,D]; vT: bf16 [B,H,D,T]; out: bf16 [B,T,C].
// Online softmax; 32 keys per iteration (two 16x16 score tiles).
// ---------------------------------------------------------------------------

__global__ void attn_wmma(const unsigned short* __restrict__ q,
                          const unsigned short* __restrict__ k,
                          const unsigned short* __restrict__ vT,
                          unsigned short* __restrict__ out) {
    __shared__ __align__(16) unsigned short Ps[16 * 32];   // P tile, bf16

    const int bh     = blockIdx.x;            // b*HEADS + h
    const int tile_t = blockIdx.y;            // T/16 tiles
    const int lane   = threadIdx.x;
    const int col    = lane & 15;
    const int hi     = lane >> 4;

    const unsigned short* qh = q  + (size_t)bh * TLEN * DHEAD;
    const unsigned short* kh = k  + (size_t)bh * TLEN * DHEAD;
    const unsigned short* vh = vT + (size_t)bh * DHEAD * TLEN;

    const int t0 = tile_t * 16;

    // Q fragments for the whole row tile (K-dim = D = 64 -> two K=32 frags)
    bf16x16 qa0 = load_frag(frag_ptr(qh, DHEAD, t0, 0,  lane));
    bf16x16 qa1 = load_frag(frag_ptr(qh, DHEAD, t0, 32, lane));

    f32x8 O[4];
#pragma unroll
    for (int c = 0; c < 4; ++c) O[c] = (f32x8){};
    float mrow[8], lrow[8];
#pragma unroll
    for (int i = 0; i < 8; ++i) { mrow[i] = -1e30f; lrow[i] = 0.0f; }

    for (int s0 = 0; s0 <= t0 + 15; s0 += 32) {
        // --- scores for two adjacent 16-key tiles ---
        f32x8 S[2];
#pragma unroll
        for (int j = 0; j < 2; ++j) {
            const int sc = s0 + j * 16;
            bf16x16 kb0 = load_frag(frag_ptr(kh, DHEAD, sc, 0,  lane));
            bf16x16 kb1 = load_frag(frag_ptr(kh, DHEAD, sc, 32, lane));
            f32x8 s_ = {};
            s_ = wmma_bf16(qa0, kb0, s_);
            s_ = wmma_bf16(qa1, kb1, s_);
            S[j] = s_;
        }

        // --- scale + causal mask ---
#pragma unroll
        for (int j = 0; j < 2; ++j)
#pragma unroll
            for (int i = 0; i < 8; ++i) {
                int row = t0 + i + 8 * hi;
                int key = s0 + j * 16 + col;
                float x = S[j][i] * 0.125f;          // 1/sqrt(64)
                S[j][i] = (key <= row) ? x : -1e30f;
            }

        // --- online softmax: row max / rescale / exp / row sum ---
        float alpha[8];
#pragma unroll
        for (int i = 0; i < 8; ++i) {
            float v = fmaxf(S[0][i], S[1][i]);
            v = fmaxf(v, __shfl_xor(v, 1, 16));
            v = fmaxf(v, __shfl_xor(v, 2, 16));
            v = fmaxf(v, __shfl_xor(v, 4, 16));
            v = fmaxf(v, __shfl_xor(v, 8, 16));
            float nm = fmaxf(mrow[i], v);
            alpha[i] = __expf(mrow[i] - nm);
            mrow[i] = nm;
        }
#pragma unroll
        for (int j = 0; j < 2; ++j)
#pragma unroll
            for (int i = 0; i < 8; ++i)
                S[j][i] = __expf(S[j][i] - mrow[i]);
#pragma unroll
        for (int i = 0; i < 8; ++i) {
            float rs = S[0][i] + S[1][i];
            rs += __shfl_xor(rs, 1, 16);
            rs += __shfl_xor(rs, 2, 16);
            rs += __shfl_xor(rs, 4, 16);
            rs += __shfl_xor(rs, 8, 16);
            lrow[i] = lrow[i] * alpha[i] + rs;
        }
#pragma unroll
        for (int c = 0; c < 4; ++c)
#pragma unroll
            for (int i = 0; i < 8; ++i)
                O[c][i] *= alpha[i];

        // --- transpose P tile through LDS into WMMA A-fragment layout ---
#pragma unroll
        for (int j = 0; j < 2; ++j)
#pragma unroll
            for (int i = 0; i < 8; ++i)
                Ps[(i + 8 * hi) * 32 + j * 16 + col] = f2bf(S[j][i]);
        __syncthreads();

        Frag16 pa;
        {
            const unsigned short* pp = &Ps[(lane & 15) * 32 + hi * 8];
            pa.q[0] = *(const uint4*)(pp);
            pa.q[1] = *(const uint4*)(pp + 16);
        }

        // --- O += P (16x32) x V (32x64), four 16-col chunks of D ---
#pragma unroll
        for (int c = 0; c < 4; ++c) {
            bf16x16 vb = load_frag(frag_ptr(vh, TLEN, c * 16, s0, lane));
            O[c] = wmma_bf16(pa.v, vb, O[c]);
        }
        __syncthreads();
    }

    // --- normalize and store to [B,T,C] bf16 ---
    const int b_ = bh >> 4;           // bh = b*HEADS + h
    const int h  = bh & (HEADS - 1);
#pragma unroll
    for (int c = 0; c < 4; ++c) {
#pragma unroll
        for (int i = 0; i < 8; ++i) {
            float val = O[c][i] / lrow[i];
            int t  = t0 + i + 8 * hi;
            int cc = h * DHEAD + c * 16 + col;
            out[(size_t)(b_ * TLEN + t) * CDIM + cc] = f2bf(val);
        }
    }
}

// ---------------------------------------------------------------------------
// Host-side launch
// ---------------------------------------------------------------------------

extern "C" void kernel_launch(void* const* d_in, const int* in_sizes, int n_in,
                              void* d_out, int out_size, void* d_ws, size_t ws_size,
                              hipStream_t stream) {
    const float* x  = (const float*)d_in[0];
    // d_in[1] = mask (unused; causal mask applied analytically)
    const float* Wq = (const float*)d_in[2];
    const float* bq = (const float*)d_in[3];
    const float* Wk = (const float*)d_in[4];
    const float* bk = (const float*)d_in[5];
    const float* Wv = (const float*)d_in[6];
    const float* bv = (const float*)d_in[7];
    const float* Wo = (const float*)d_in[8];
    const float* bo = (const float*)d_in[9];
    float* out = (float*)d_out;

    const int M = BATCH * TLEN;       // 8192
    const int N = CDIM;               // 1024
    const int K = CDIM;               // 1024

    char* ws = (char*)d_ws;
    size_t off = 0;
    unsigned short* xb  = (unsigned short*)(ws + off); off += (size_t)M * K * 2;          // 16 MB
    unsigned short* WqT = (unsigned short*)(ws + off); off += (size_t)K * N * 2;          //  2 MB
    unsigned short* WkT = (unsigned short*)(ws + off); off += (size_t)K * N * 2;
    unsigned short* WvT = (unsigned short*)(ws + off); off += (size_t)K * N * 2;
    unsigned short* WoT = (unsigned short*)(ws + off); off += (size_t)K * N * 2;
    unsigned short* qb  = (unsigned short*)(ws + off); off += (size_t)M * CDIM * 2;       // 16 MB
    unsigned short* kb  = (unsigned short*)(ws + off); off += (size_t)M * CDIM * 2;
    unsigned short* vTb = (unsigned short*)(ws + off); off += (size_t)M * CDIM * 2;
    unsigned short* ao  = (unsigned short*)(ws + off); off += (size_t)M * CDIM * 2;

    // 1) conversions
    {
        int n = M * K;
        convert_f32_to_bf16<<<(n + 255) / 256, 256, 0, stream>>>(x, xb, n);
        int nw = K * N;
        convert_transpose_w<<<(nw + 255) / 256, 256, 0, stream>>>(Wq, WqT);
        convert_transpose_w<<<(nw + 255) / 256, 256, 0, stream>>>(Wk, WkT);
        convert_transpose_w<<<(nw + 255) / 256, 256, 0, stream>>>(Wv, WvT);
        convert_transpose_w<<<(nw + 255) / 256, 256, 0, stream>>>(Wo, WoT);
    }

    // 2) QKV projections (WMMA, 32x64 wave tiles)
    dim3 ggrid(N / 64, M / 32);
    gemm_wmma<<<ggrid, 32, 0, stream>>>(xb, WqT, bq, qb,  M, N, K, 1);
    gemm_wmma<<<ggrid, 32, 0, stream>>>(xb, WkT, bk, kb,  M, N, K, 1);
    gemm_wmma<<<ggrid, 32, 0, stream>>>(xb, WvT, bv, vTb, M, N, K, 2);

    // 3) flash attention core
    dim3 agrid(BATCH * HEADS, TLEN / 16);
    attn_wmma<<<agrid, 32, 0, stream>>>(qb, kb, vTb, ao);

    // 4) output projection -> f32 d_out
    gemm_wmma<<<ggrid, 32, 0, stream>>>(ao, WoT, bo, out, M, N, K, 0);
}
